// Decoder_12970801234218
// MI455X (gfx1250) — matrix operand order repaired and checked by reading, not compiled, not run on hardware
//
#include <hip/hip_runtime.h>
#include <hip/hip_bf16.h>

// ---------------------------------------------------------------------------
// Pointer-network decoder for MI455X (gfx1250, wave32, WMMA).
// GEMMs: v_wmma_f32_16x16x32_bf16, 8 waves/block sharing an async-to-LDS
// double-buffered B strip. e_g/e_p precomputed as ONE [65536,512]x[512,512]
// GEMM each, stored bf16 so both stay resident in the 192MB L2 for the scan.
// ---------------------------------------------------------------------------

#define B_   128
#define L_   512
#define E_   512
#define H_   512
#define H4_  2048
#define T_   64
#define NEGV (-1000000000.0f)
#define CEXP 10.0f
#define GROWB 40   /* LDS row pitch (elements) for 32-element B rows: 80B, 16B-aligned, 20-bank stride */

typedef __bf16 bf16_t;
typedef bf16_t        v16bf __attribute__((ext_vector_type(16)));
typedef unsigned short v16us __attribute__((ext_vector_type(16)));
typedef unsigned short v8us  __attribute__((ext_vector_type(8)));
typedef float          v8f   __attribute__((ext_vector_type(8)));
typedef int            v4i   __attribute__((ext_vector_type(4)));

__device__ __forceinline__ unsigned short f2bf(float x) {
  unsigned int u = __float_as_uint(x);
  unsigned int r = (u + 0x7FFFu + ((u >> 16) & 1u)) >> 16;   // RNE
  return (unsigned short)r;
}
__device__ __forceinline__ float bf2f(unsigned short h) {
  return __uint_as_float(((unsigned int)h) << 16);
}
__device__ __forceinline__ float sigf(float x) { return 1.0f / (1.0f + __expf(-x)); }

// 16-byte global -> LDS copy. Prefer gfx1250 async-to-LDS (ASYNCcnt) path.
__device__ __forceinline__ void stage16(const unsigned short* g, unsigned short* l) {
#if __has_builtin(__builtin_amdgcn_global_load_async_to_lds_b128)
  typedef __attribute__((address_space(1))) v4i* as1_v4i;
  typedef __attribute__((address_space(3))) v4i* as3_v4i;
  __builtin_amdgcn_global_load_async_to_lds_b128(
      (as1_v4i)(v4i*)const_cast<unsigned short*>(g),
      (as3_v4i)(v4i*)l, 0, 0);
#else
  *reinterpret_cast<v8us*>(l) = *reinterpret_cast<const v8us*>(g);
#endif
}
__device__ __forceinline__ void stage_wait_and_barrier() {
#if __has_builtin(__builtin_amdgcn_global_load_async_to_lds_b128)
#if __has_builtin(__builtin_amdgcn_s_wait_asynccnt)
  __builtin_amdgcn_s_wait_asynccnt(0);
#else
  asm volatile("s_wait_asynccnt 0x0" ::: "memory");
#endif
#endif
  __syncthreads();
}

// ---------------------------------------------------------------------------
// bf16 WMMA GEMM: C[M,N] = A[M,K] @ Bt[N,K]^T + bias (+ C if acc)
// Block = 256 threads = 8 waves -> 128x64 output tile. The 64x32 B-strip per
// k-step is staged into LDS (double-buffered, async) and shared by all waves;
// each wave owns one 16-row A block and does 4 WMMAs per k-step.
// Requires M%128==0, N%64==0, K%32==0.
// ---------------------------------------------------------------------------
__global__ void __launch_bounds__(256)
k_gemm_bf16(const unsigned short* __restrict__ A,
            const unsigned short* __restrict__ Bt,
            const float* __restrict__ bias,
            float* __restrict__ Cf,
            unsigned short* __restrict__ Cbf,
            int M, int N, int K, int accumulate)
{
  __shared__ __align__(16) unsigned short bsh[2][64 * GROWB];

  const int tid  = (int)threadIdx.x;
  const int lane = tid & 31;
  const int wave = tid >> 5;
  const int blocksM = M >> 7;                 // M / 128
  const int bm  = (int)blockIdx.x % blocksM;
  const int tn4 = (int)blockIdx.x / blocksM;

  const int tm   = bm * 8 + wave;
  const int arow = tm * 16 + (lane & 15);
  const int ac0  = (lane >> 4) * 16;          // A: elems 0..15 -> K=c0..c0+15
  const int nloc = lane & 15;                 // B col within a 16-tile
  const int bk0  = (lane >> 4) * 16;          // B: elems 0..15 -> K=bk0..bk0+15
  const int n0   = tn4 * 64;

  // cooperative staging: thread copies row sn (of 64), 16-byte chunk sc (of 4)
  const int sn = tid >> 2;
  const int sc = tid & 3;
  const unsigned short* gsrc = Bt + (size_t)(n0 + sn) * K + sc * 8;
  unsigned short* lrow = &bsh[0][0] + sn * GROWB + sc * 8;

  v8f acc[4];
#pragma unroll
  for (int t = 0; t < 4; ++t) acc[t] = (v8f)0.0f;

  const unsigned short* arow_p = A + (size_t)arow * K;

  // prologue: stage k-step 0 into buffer 0
  stage16(gsrc, lrow);
  stage_wait_and_barrier();

  for (int kb = 0; kb < K; kb += 32) {
    const int buf = (kb >> 5) & 1;
    if (kb + 32 < K)                           // prefetch next strip into other buffer
      stage16(gsrc + kb + 32, lrow + (buf ^ 1) * (64 * GROWB));

    v8us alo = *reinterpret_cast<const v8us*>(arow_p + kb + ac0);
    v8us ahi = *reinterpret_cast<const v8us*>(arow_p + kb + ac0 + 8);
    v16us af = __builtin_shufflevector(alo, ahi, 0,1,2,3,4,5,6,7,8,9,10,11,12,13,14,15);

#pragma unroll
    for (int t = 0; t < 4; ++t) {
      v16us bfv = *reinterpret_cast<const v16us*>(&bsh[buf][(t * 16 + nloc) * GROWB + bk0]);
      acc[t] = __builtin_amdgcn_wmma_f32_16x16x32_bf16(
          false, __builtin_bit_cast(v16bf, af),
          false, __builtin_bit_cast(v16bf, bfv),
          (short)0, acc[t], false, false);
    }
    stage_wait_and_barrier();                  // next strip ready; all waves done with cur
  }

  // C/D layout: VGPR r, lanes 0-15 -> M=r, lanes 16-31 -> M=r+8; N = lane%16.
  const int rofs = (lane >> 4) << 3;
#pragma unroll
  for (int t = 0; t < 4; ++t) {
    const int col = n0 + t * 16 + (lane & 15);
    const float bv = bias ? bias[col] : 0.0f;
#pragma unroll
    for (int r = 0; r < 8; ++r) {
      const int row = tm * 16 + r + rofs;
      float v = acc[t][r] + bv;
      if (accumulate) v += Cf[(size_t)row * N + col];
      if (Cbf) Cbf[(size_t)row * N + col] = f2bf(v);
      else     Cf [(size_t)row * N + col] = v;
    }
  }
}

// ---------------------------------------------------------------------------
// Conversions / init
// ---------------------------------------------------------------------------
__global__ void k_cvt(const float* __restrict__ src, unsigned short* __restrict__ dst, int n) {
  for (int i = blockIdx.x * blockDim.x + threadIdx.x; i < n; i += gridDim.x * blockDim.x)
    dst[i] = f2bf(src[i]);
}

// dst[N,K] (bf16) = transpose(src[K,N] f32)
__global__ void k_cvt_T(const float* __restrict__ src, unsigned short* __restrict__ dst,
                        int K, int N) {
  int total = K * N;
  for (int i = blockIdx.x * blockDim.x + threadIdx.x; i < total; i += gridDim.x * blockDim.x) {
    int nn = i / K;
    int kk = i - nn * K;
    dst[i] = f2bf(src[(size_t)kk * N + nn]);
  }
}

__global__ void k_init(const float* __restrict__ h0, const float* __restrict__ c0,
                       const float* __restrict__ x0,
                       float* __restrict__ hx, float* __restrict__ cx,
                       float* __restrict__ xcur, int* __restrict__ mask,
                       int* __restrict__ prev, unsigned short* __restrict__ h_bf) {
  int i = blockIdx.x * blockDim.x + threadIdx.x;
  if (i < B_ * L_) mask[i] = 0;
  if (i < B_)      prev[i] = 0;
  if (i < B_ * H_) { hx[i] = h0[i]; cx[i] = c0[i]; h_bf[i] = f2bf(h0[i]); }
  if (i < B_ * E_) xcur[i] = x0[i];
}

__global__ void k_mask_update(int* __restrict__ mask, const int* __restrict__ prev) {
  int b = threadIdx.x;
  if (b < B_) mask[b * L_ + prev[b]] = 1;
}

// ---------------------------------------------------------------------------
// LSTM cell elementwise
// ---------------------------------------------------------------------------
__global__ void k_lstm(const float* __restrict__ gates, float* __restrict__ hx,
                       float* __restrict__ cx, unsigned short* __restrict__ h_bf) {
  int idx = blockIdx.x * blockDim.x + threadIdx.x;
  if (idx >= B_ * H_) return;
  int b = idx >> 9, h = idx & (H_ - 1);
  const float* g = gates + (size_t)b * H4_;
  float ig = g[h], fg = g[H_ + h], gg = g[2 * H_ + h], og = g[3 * H_ + h];
  float cy = sigf(fg) * cx[idx] + sigf(ig) * tanhf(gg);
  float hy = sigf(og) * tanhf(cy);
  cx[idx] = cy; hx[idx] = hy; h_bf[idx] = f2bf(hy);
}

// ---------------------------------------------------------------------------
// u[b,l] = sum_h v[h] * tanh(q[b,h] + E[(l,b),h])   (one wave per (b,l))
// ---------------------------------------------------------------------------
__global__ void k_attn_scores(const float* __restrict__ q,
                              const unsigned short* __restrict__ Eref,
                              const float* __restrict__ vvec,
                              float* __restrict__ u) {
  int wave = (int)((blockIdx.x * blockDim.x + threadIdx.x) >> 5);
  int lane = (int)(threadIdx.x & 31);
  if (wave >= B_ * L_) return;
  int b = wave >> 9;           // / L_
  int l = wave & (L_ - 1);
  const unsigned short* e = Eref + ((size_t)l * B_ + b) * H_;
  const float* qb = q + (size_t)b * H_;
  float acc = 0.0f;
  for (int h = lane; h < H_; h += 32)
    acc += vvec[h] * tanhf(qb[h] + bf2f(e[h]));
#pragma unroll
  for (int off = 16; off > 0; off >>= 1) acc += __shfl_xor(acc, off, 32);
  if (lane == 0) u[(size_t)b * L_ + l] = acc;
}

// ---------------------------------------------------------------------------
// masked softmax over L (block per b, 512 threads)
// ---------------------------------------------------------------------------
__global__ void k_softmax(const float* __restrict__ u, const int* __restrict__ mask,
                          float* __restrict__ a) {
  __shared__ float red[L_];
  int b = blockIdx.x, l = threadIdx.x;
  float logit = mask[b * L_ + l] ? NEGV : u[b * L_ + l];
  red[l] = logit; __syncthreads();
  for (int s = L_ / 2; s > 0; s >>= 1) { if (l < s) red[l] = fmaxf(red[l], red[l + s]); __syncthreads(); }
  float mx = red[0]; __syncthreads();
  float ex = __expf(logit - mx);
  red[l] = ex; __syncthreads();
  for (int s = L_ / 2; s > 0; s >>= 1) { if (l < s) red[l] += red[l + s]; __syncthreads(); }
  a[b * L_ + l] = ex / red[0];
}

// ---------------------------------------------------------------------------
// g_l[b,h] = sum_l E[(l,b),h] * a[b,l]    (thread per (b,h))
// ---------------------------------------------------------------------------
__global__ void k_glimpse(const unsigned short* __restrict__ Eg, const float* __restrict__ a,
                          float* __restrict__ gl, unsigned short* __restrict__ gl_bf) {
  int idx = blockIdx.x * blockDim.x + threadIdx.x;
  if (idx >= B_ * H_) return;
  int b = idx >> 9, h = idx & (H_ - 1);
  const float* ab = a + (size_t)b * L_;
  float acc = 0.0f;
  for (int l = 0; l < L_; ++l)
    acc += bf2f(Eg[((size_t)l * B_ + b) * H_ + h]) * ab[l];
  gl[idx] = acc; gl_bf[idx] = f2bf(acc);
}

// ---------------------------------------------------------------------------
// pointer head: logits -> probs (to d_out), argmax -> prev/sels, gather new x
// ---------------------------------------------------------------------------
__global__ void k_pointer_out(const float* __restrict__ up, const int* __restrict__ mask,
                              const float* __restrict__ emb,      // [L,B,E]
                              float* __restrict__ probs_out,      // [B,L]
                              float* __restrict__ sel_out,        // [B]
                              int* __restrict__ prev, float* __restrict__ xcur) {
  __shared__ float red[L_];
  __shared__ int  redi[L_];
  __shared__ int  sel_sh;
  int b = blockIdx.x, l = threadIdx.x;
  float logit = mask[b * L_ + l] ? NEGV : CEXP * tanhf(up[b * L_ + l]);
  red[l] = logit; __syncthreads();
  for (int s = L_ / 2; s > 0; s >>= 1) { if (l < s) red[l] = fmaxf(red[l], red[l + s]); __syncthreads(); }
  float mx = red[0]; __syncthreads();
  float ex = __expf(logit - mx);
  red[l] = ex; __syncthreads();
  for (int s = L_ / 2; s > 0; s >>= 1) { if (l < s) red[l] += red[l + s]; __syncthreads(); }
  probs_out[b * L_ + l] = ex / red[0];
  __syncthreads();
  // argmax(probs) == argmax(logits); first index on ties (matches jnp.argmax)
  red[l] = logit; redi[l] = l; __syncthreads();
  for (int s = L_ / 2; s > 0; s >>= 1) {
    if (l < s) {
      float v2 = red[l + s]; int i2 = redi[l + s];
      if (v2 > red[l] || (v2 == red[l] && i2 < redi[l])) { red[l] = v2; redi[l] = i2; }
    }
    __syncthreads();
  }
  if (l == 0) { sel_sh = redi[0]; prev[b] = redi[0]; sel_out[b] = (float)redi[0]; }
  __syncthreads();
  int idx = sel_sh;
  for (int e = l; e < E_; e += L_)
    xcur[(size_t)b * E_ + e] = emb[((size_t)idx * B_ + b) * E_ + e];
}

__global__ void k_final(const float* __restrict__ hx, const float* __restrict__ cx,
                        float* __restrict__ out) {
  int i = blockIdx.x * blockDim.x + threadIdx.x;
  if (i < B_ * H_) { out[i] = hx[i]; out[B_ * H_ + i] = cx[i]; }
}

// ---------------------------------------------------------------------------
extern "C" void kernel_launch(void* const* d_in, const int* in_sizes, int n_in,
                              void* d_out, int out_size, void* d_ws, size_t ws_size,
                              hipStream_t stream) {
  (void)in_sizes; (void)n_in; (void)out_size; (void)ws_size;
  const float* decoder_input = (const float*)d_in[0];
  const float* emb     = (const float*)d_in[1];
  const float* h0      = (const float*)d_in[2];
  const float* c0      = (const float*)d_in[3];
  const float* context = (const float*)d_in[4];
  /* d_in[5] = max_length (device scalar); fixed T_=64 per setup */
  const float* W_in  = (const float*)d_in[6];
  const float* b_in  = (const float*)d_in[7];
  const float* W_h   = (const float*)d_in[8];
  const float* b_h   = (const float*)d_in[9];
  const float* g_Wq  = (const float*)d_in[10];
  const float* g_bq  = (const float*)d_in[11];
  const float* g_Wref= (const float*)d_in[12];
  const float* g_bref= (const float*)d_in[13];
  const float* g_v   = (const float*)d_in[14];
  const float* p_Wq  = (const float*)d_in[15];
  const float* p_bq  = (const float*)d_in[16];
  const float* p_Wref= (const float*)d_in[17];
  const float* p_bref= (const float*)d_in[18];
  const float* p_v   = (const float*)d_in[19];

  char* ws = (char*)d_ws;
  size_t off = 0;
  auto alloc = [&](size_t bytes) -> void* {
    void* p = ws + off; off += (bytes + 255) & ~(size_t)255; return p;
  };
  unsigned short* ctx_bf = (unsigned short*)alloc((size_t)L_ * B_ * H_ * 2);
  unsigned short* Eg     = (unsigned short*)alloc((size_t)L_ * B_ * H_ * 2);
  unsigned short* Ep     = (unsigned short*)alloc((size_t)L_ * B_ * H_ * 2);
  unsigned short* WinT   = (unsigned short*)alloc((size_t)H4_ * E_ * 2);
  unsigned short* WhT    = (unsigned short*)alloc((size_t)H4_ * H_ * 2);
  unsigned short* gWqT   = (unsigned short*)alloc((size_t)H_ * H_ * 2);
  unsigned short* pWqT   = (unsigned short*)alloc((size_t)H_ * H_ * 2);
  unsigned short* gWrefB = (unsigned short*)alloc((size_t)H_ * H_ * 2);
  unsigned short* pWrefB = (unsigned short*)alloc((size_t)H_ * H_ * 2);
  float* hx    = (float*)alloc((size_t)B_ * H_ * 4);
  float* cx    = (float*)alloc((size_t)B_ * H_ * 4);
  float* xcur  = (float*)alloc((size_t)B_ * E_ * 4);
  float* gates = (float*)alloc((size_t)B_ * H4_ * 4);
  float* qg    = (float*)alloc((size_t)B_ * H_ * 4);
  float* qp    = (float*)alloc((size_t)B_ * H_ * 4);
  float* gl    = (float*)alloc((size_t)B_ * H_ * 4);
  float* ug    = (float*)alloc((size_t)B_ * L_ * 4);
  float* av    = (float*)alloc((size_t)B_ * L_ * 4);
  float* upb   = (float*)alloc((size_t)B_ * L_ * 4);
  unsigned short* x_bf  = (unsigned short*)alloc((size_t)B_ * E_ * 2);
  unsigned short* h_bf  = (unsigned short*)alloc((size_t)B_ * H_ * 2);
  unsigned short* gl_bf = (unsigned short*)alloc((size_t)B_ * H_ * 2);
  int* mask = (int*)alloc((size_t)B_ * L_ * 4);
  int* prev = (int*)alloc((size_t)B_ * 4);

  auto gemm = [&](const unsigned short* A, const unsigned short* Bt, const float* bias,
                  float* Cf, unsigned short* Cbf, int M, int N, int K, int acc) {
    int blocks = (M >> 7) * (N >> 6);          // block = 128x64 tile, 8 waves
    k_gemm_bf16<<<blocks, 256, 0, stream>>>(A, Bt, bias, Cf, Cbf, M, N, K, acc);
  };

  // ---- precompute: conversions + e_g / e_p as single big GEMMs ------------
  k_cvt  <<<1024, 256, 0, stream>>>(context, ctx_bf, L_ * B_ * H_);
  k_cvt  <<<256,  256, 0, stream>>>(g_Wref, gWrefB, H_ * H_);
  k_cvt  <<<256,  256, 0, stream>>>(p_Wref, pWrefB, H_ * H_);
  k_cvt_T<<<1024, 256, 0, stream>>>(W_in, WinT, E_, H4_);
  k_cvt_T<<<1024, 256, 0, stream>>>(W_h,  WhT,  H_, H4_);
  k_cvt_T<<<256,  256, 0, stream>>>(g_Wq, gWqT, H_, H_);
  k_cvt_T<<<256,  256, 0, stream>>>(p_Wq, pWqT, H_, H_);
  gemm(ctx_bf, gWrefB, g_bref, nullptr, Eg, L_ * B_, H_, H_, 0);
  gemm(ctx_bf, pWrefB, p_bref, nullptr, Ep, L_ * B_, H_, H_, 0);
  k_init<<<256, 256, 0, stream>>>(h0, c0, decoder_input, hx, cx, xcur, mask, prev, h_bf);

  float* probs_base = (float*)d_out;                           // [T,B,L]
  float* sels_base  = probs_base + (size_t)T_ * B_ * L_;       // [T,B] (as float)
  float* hx_out     = sels_base + (size_t)T_ * B_;             // [B,H] + [B,H]

  // ---- sequential decode ---------------------------------------------------
  for (int t = 0; t < T_; ++t) {
    if (t > 0) k_mask_update<<<1, 128, 0, stream>>>(mask, prev);
    k_cvt<<<256, 256, 0, stream>>>(xcur, x_bf, B_ * E_);
    gemm(x_bf, WinT, b_in, gates, nullptr, B_, H4_, E_, 0);
    gemm(h_bf, WhT,  b_h,  gates, nullptr, B_, H4_, H_, 1);
    k_lstm<<<256, 256, 0, stream>>>(gates, hx, cx, h_bf);
    gemm(h_bf, gWqT, g_bq, qg, nullptr, B_, H_, H_, 0);
    k_attn_scores<<<(B_ * L_) / 8, 256, 0, stream>>>(qg, Eg, g_v, ug);
    k_softmax<<<B_, L_, 0, stream>>>(ug, mask, av);
    k_glimpse<<<256, 256, 0, stream>>>(Eg, av, gl, gl_bf);
    gemm(gl_bf, pWqT, p_bq, qp, nullptr, B_, H_, H_, 0);
    k_attn_scores<<<(B_ * L_) / 8, 256, 0, stream>>>(qp, Ep, p_v, upb);
    k_pointer_out<<<B_, L_, 0, stream>>>(upb, mask, emb,
                                         probs_base + (size_t)t * B_ * L_,
                                         sels_base + (size_t)t * B_,
                                         prev, xcur);
  }
  k_final<<<256, 256, 0, stream>>>(hx, cx, hx_out);
}
